// FocalNet_17935783428721
// MI455X (gfx1250) — matrix operand order repaired and checked by reading, not compiled, not run on hardware
//
#include <hip/hip_runtime.h>
#include <hip/hip_bf16.h>
#include <math.h>

typedef __bf16 bf16;
typedef __attribute__((ext_vector_type(16))) bf16 v16bf;
typedef __attribute__((ext_vector_type(8)))  bf16 v8bf;
typedef __attribute__((ext_vector_type(8)))  float v8f;

#define CDIM 128
#define HH 56
#define WW 56
#define BB 64
#define LL (HH*WW)                 // 3136
#define NROWS ((size_t)BB*LL)      // 200704

// ---------- helpers ----------
__device__ __forceinline__ unsigned short f2bf(float f) {
    unsigned u = __float_as_uint(f);
    u += 0x7fffu + ((u >> 16) & 1u);   // round-to-nearest-even
    return (unsigned short)(u >> 16);
}

__device__ __forceinline__ float geluf(float x) {
    return 0.5f * x * (1.0f + erff(x * 0.70710678118654752f));
}

// ---------- weight transpose + bf16 convert: Wt[n*K+k] = W[k*Nsrc+n] ----------
__global__ void k_w2bf_t(const float* __restrict__ W, unsigned short* __restrict__ Wt,
                         int K, int Nsrc, int Npad) {
    int idx = blockIdx.x * blockDim.x + threadIdx.x;
    if (idx >= Npad * K) return;
    int n = idx / K, k = idx - n * K;
    float v = (n < Nsrc) ? W[(size_t)k * Nsrc + n] : 0.0f;
    Wt[idx] = f2bf(v);
}

// ---------- LayerNorm (over C=128) -> bf16 ; one wave per row ----------
__global__ void k_ln_bf16(const float* __restrict__ x, const float* __restrict__ g,
                          const float* __restrict__ b, unsigned short* __restrict__ y,
                          long long rows) {
    long long wid = ((long long)blockIdx.x * blockDim.x + threadIdx.x) >> 5;
    int lane = threadIdx.x & 31;
    if (wid >= rows) return;
    const float* row = x + (size_t)wid * CDIM;
    float4 v = ((const float4*)row)[lane];
    float s = v.x + v.y + v.z + v.w;
    for (int o = 16; o > 0; o >>= 1) s += __shfl_xor(s, o, 32);
    float mu = s * (1.0f / CDIM);
    float d0 = v.x - mu, d1 = v.y - mu, d2 = v.z - mu, d3 = v.w - mu;
    float ss = d0 * d0 + d1 * d1 + d2 * d2 + d3 * d3;
    for (int o = 16; o > 0; o >>= 1) ss += __shfl_xor(ss, o, 32);
    float inv = rsqrtf(ss * (1.0f / CDIM) + 1e-5f);
    int c = lane * 4;
    unsigned short* yr = y + (size_t)wid * CDIM + c;
    yr[0] = f2bf(d0 * inv * g[c + 0] + b[c + 0]);
    yr[1] = f2bf(d1 * inv * g[c + 1] + b[c + 1]);
    yr[2] = f2bf(d2 * inv * g[c + 2] + b[c + 2]);
    yr[3] = f2bf(d3 * inv * g[c + 3] + b[c + 3]);
}

// ---------- depthwise conv + GELU + gated accumulate ----------
__global__ void k_dwconv(const float* __restrict__ in, const float* __restrict__ kw,
                         const float* __restrict__ gates, float* __restrict__ out,
                         float* __restrict__ ctx_all, int ksz, int gidx, int first) {
    size_t idx = (size_t)blockIdx.x * blockDim.x + threadIdx.x;
    if (idx >= NROWS * CDIM) return;
    int c = (int)(idx & (CDIM - 1));
    size_t p = idx >> 7;
    int w = (int)(p % WW);
    size_t t = p / WW;
    int h = (int)(t % HH);
    int b = (int)(t / HH);
    int r = ksz >> 1;
    float acc = 0.0f;
    for (int dy = 0; dy < ksz; dy++) {
        int hy = h + dy - r;
        if ((unsigned)hy >= (unsigned)HH) continue;
        for (int dxp = 0; dxp < ksz; dxp++) {
            int wx = w + dxp - r;
            if ((unsigned)wx >= (unsigned)WW) continue;
            acc += in[(((size_t)b * HH + hy) * WW + wx) * CDIM + c] *
                   kw[(dy * ksz + dxp) * CDIM + c];
        }
    }
    float gv = geluf(acc);
    out[idx] = gv;
    float ga = gates[p * 4 + gidx];
    if (first) ctx_all[idx] = gv * ga;
    else       ctx_all[idx] += gv * ga;
}

// ---------- global mean over (H,W) + GELU : one block per (b,c) ----------
__global__ void k_pool(const float* __restrict__ ctx, float* __restrict__ g) {
    __shared__ float sm[256];
    int b = blockIdx.x >> 7;
    int c = blockIdx.x & 127;
    float s = 0.0f;
    for (int p = threadIdx.x; p < LL; p += 256)
        s += ctx[((size_t)b * LL + p) * CDIM + c];
    sm[threadIdx.x] = s;
    __syncthreads();
    for (int o = 128; o > 0; o >>= 1) {
        if (threadIdx.x < o) sm[threadIdx.x] += sm[threadIdx.x + o];
        __syncthreads();
    }
    if (threadIdx.x == 0) g[blockIdx.x] = geluf(sm[0] * (1.0f / LL));
}

// ---------- ctx_all += ctx_global * gate3 ; convert to bf16 ----------
__global__ void k_finalize(const float* __restrict__ ctx_all, const float* __restrict__ g,
                           const float* __restrict__ gates, unsigned short* __restrict__ obf) {
    size_t idx = (size_t)blockIdx.x * blockDim.x + threadIdx.x;
    if (idx >= NROWS * CDIM) return;
    int c = (int)(idx & 127);
    size_t p = idx >> 7;
    int b = (int)(p / LL);
    float v = ctx_all[idx] + g[b * CDIM + c] * gates[p * 4 + 3];
    obf[idx] = f2bf(v);
}

// ---------- bf16 WMMA GEMM: 32(M) x 64(N) per wave (2 m-tiles x 4 n-tiles) ----------
// KTOT compile-time: all inter-row strides become immediate offsets (2 base pointers only).
// A: [M x KTOT] bf16 row-major ; Bt: [Npad x KTOT] bf16 row-major (pre-transposed weights)
// M multiple of 32 ; NT (n-tiles of 16) multiple of 4.
// MODE 0: split write q/ctx/gates (Npad=320, bias valid < 260)
// MODE 1: modulator = acc+bias ; outbf = bf16(aux(q) * modulator)
// MODE 2: outf = aux(shortcut) + acc + bias          (x1)
// MODE 3: outbf = bf16(gelu(acc+bias)), ld 512       (fc1 hidden)
// MODE 4: outf = aux(x1) + acc + bias                (final output)
template<int KTOT, int MODE>
__global__ void k_gemm(const unsigned short* __restrict__ A,
                       const unsigned short* __restrict__ Bt,
                       const float* __restrict__ bias,
                       int NT, long long totalGroups,
                       const float* __restrict__ aux,
                       float* __restrict__ outf,
                       float* __restrict__ out1,
                       float* __restrict__ out2,
                       unsigned short* __restrict__ outbf) {
    long long grp = (long long)blockIdx.x * 8 + (threadIdx.x >> 5);
    if (grp >= totalGroups) return;
    int lane = threadIdx.x & 31;
    int ngroups = NT >> 2;                       // groups of 4 n-tiles
    int mt = (int)(grp / ngroups);
    int ng = (int)(grp - (long long)mt * ngroups);
    int M0 = mt * 32, N0 = ng * 64;
    int half = lane >> 4, lm = lane & 15;

    constexpr int R16 = 16 * KTOT;               // element stride of 16 rows

    // Single base pointer per matrix; all other fragments at compile-time immediate offsets.
    const unsigned short* abase = A  + (size_t)(M0 + lm) * KTOT + half * 16;
    const unsigned short* bbase = Bt + (size_t)(N0 + lm) * KTOT + half * 16;

    v8f acc00 = {}, acc01 = {}, acc02 = {}, acc03 = {};
    v8f acc10 = {}, acc11 = {}, acc12 = {}, acc13 = {};
#pragma unroll 4
    for (int k0 = 0; k0 < KTOT; k0 += 32) {
        v8bf a0lo = *(const v8bf*)(abase + k0);
        v8bf a0hi = *(const v8bf*)(abase + k0 + 8);
        v8bf a1lo = *(const v8bf*)(abase + R16 + k0);
        v8bf a1hi = *(const v8bf*)(abase + R16 + k0 + 8);
        v16bf a0, a1;
#pragma unroll
        for (int i = 0; i < 8; i++) {
            a0[i] = a0lo[i]; a0[i + 8] = a0hi[i];
            a1[i] = a1lo[i]; a1[i + 8] = a1hi[i];
        }
        v16bf b0 = *(const v16bf*)(bbase + 0 * R16 + k0);
        v16bf b1 = *(const v16bf*)(bbase + 1 * R16 + k0);
        v16bf b2 = *(const v16bf*)(bbase + 2 * R16 + k0);
        v16bf b3 = *(const v16bf*)(bbase + 3 * R16 + k0);
        acc00 = __builtin_amdgcn_wmma_f32_16x16x32_bf16(false, a0, false, b0, (short)0, acc00, false, false);
        acc01 = __builtin_amdgcn_wmma_f32_16x16x32_bf16(false, a0, false, b1, (short)0, acc01, false, false);
        acc02 = __builtin_amdgcn_wmma_f32_16x16x32_bf16(false, a0, false, b2, (short)0, acc02, false, false);
        acc03 = __builtin_amdgcn_wmma_f32_16x16x32_bf16(false, a0, false, b3, (short)0, acc03, false, false);
        acc10 = __builtin_amdgcn_wmma_f32_16x16x32_bf16(false, a1, false, b0, (short)0, acc10, false, false);
        acc11 = __builtin_amdgcn_wmma_f32_16x16x32_bf16(false, a1, false, b1, (short)0, acc11, false, false);
        acc12 = __builtin_amdgcn_wmma_f32_16x16x32_bf16(false, a1, false, b2, (short)0, acc12, false, false);
        acc13 = __builtin_amdgcn_wmma_f32_16x16x32_bf16(false, a1, false, b3, (short)0, acc13, false, false);
    }

#pragma unroll
    for (int i = 0; i < 2; i++) {
        int Mbase = M0 + i * 16;
#pragma unroll
        for (int j = 0; j < 4; j++) {
            v8f acc;
            if (i == 0) acc = (j == 0) ? acc00 : (j == 1) ? acc01 : (j == 2) ? acc02 : acc03;
            else        acc = (j == 0) ? acc10 : (j == 1) ? acc11 : (j == 2) ? acc12 : acc13;
            int Ncol = N0 + j * 16 + lm;
#pragma unroll
            for (int r = 0; r < 8; r++) {
                int Mr = Mbase + r + half * 8;   // C/D layout: vgpr r -> M=r (+8 for upper half-wave)
                float v = acc[r];
                if constexpr (MODE == 0) {
                    if (Ncol < 260) v += bias[Ncol];
                    if (Ncol < 128)      outf[(size_t)Mr * CDIM + Ncol] = v;          // q
                    else if (Ncol < 256) out1[(size_t)Mr * CDIM + (Ncol - 128)] = v;  // ctx
                    else if (Ncol < 260) out2[(size_t)Mr * 4 + (Ncol - 256)] = v;     // gates
                } else if constexpr (MODE == 1) {
                    v += bias[Ncol];
                    float xm = aux[(size_t)Mr * CDIM + Ncol] * v;
                    outbf[(size_t)Mr * CDIM + Ncol] = f2bf(xm);
                } else if constexpr (MODE == 2) {
                    v += bias[Ncol];
                    outf[(size_t)Mr * CDIM + Ncol] = aux[(size_t)Mr * CDIM + Ncol] + v;
                } else if constexpr (MODE == 3) {
                    v = geluf(v + bias[Ncol]);
                    outbf[(size_t)Mr * 512 + Ncol] = f2bf(v);
                } else {
                    v += bias[Ncol];
                    outf[(size_t)Mr * CDIM + Ncol] = aux[(size_t)Mr * CDIM + Ncol] + v;
                }
            }
        }
    }
}

// ---------- host ----------
extern "C" void kernel_launch(void* const* d_in, const int* in_sizes, int n_in,
                              void* d_out, int out_size, void* d_ws, size_t ws_size,
                              hipStream_t stream) {
    const float* x      = (const float*)d_in[0];
    const float* f_w    = (const float*)d_in[1];
    const float* f_b    = (const float*)d_in[2];
    const float* k0w    = (const float*)d_in[3];
    const float* k1w    = (const float*)d_in[4];
    const float* k2w    = (const float*)d_in[5];
    const float* h_w    = (const float*)d_in[6];
    const float* h_b    = (const float*)d_in[7];
    const float* proj_w = (const float*)d_in[8];
    const float* proj_b = (const float*)d_in[9];
    const float* ln1_g  = (const float*)d_in[10];
    const float* ln1_b  = (const float*)d_in[11];
    const float* ln2_g  = (const float*)d_in[12];
    const float* ln2_b  = (const float*)d_in[13];
    const float* fc1_w  = (const float*)d_in[14];
    const float* fc1_b  = (const float*)d_in[15];
    const float* fc2_w  = (const float*)d_in[16];
    const float* fc2_b  = (const float*)d_in[17];
    float* out = (float*)d_out;

    const size_t N = NROWS;
    char* ws = (char*)d_ws;
    size_t off = 0;
    auto alloc = [&](size_t bytes) -> void* {
        void* p = ws + off;
        off = (off + bytes + 255) & ~(size_t)255;
        return p;
    };

    unsigned short* f_wt   = (unsigned short*)alloc(320 * 128 * 2);  // padded to 320 rows
    unsigned short* h_wt   = (unsigned short*)alloc(128 * 128 * 2);
    unsigned short* p_wt   = (unsigned short*)alloc(128 * 128 * 2);
    unsigned short* fc1_wt = (unsigned short*)alloc(512 * 128 * 2);
    unsigned short* fc2_wt = (unsigned short*)alloc(128 * 512 * 2);
    float*          gbuf   = (float*)alloc(BB * CDIM * 4);
    float*          q      = (float*)alloc(N * CDIM * 4);
    float*          ctxA   = (float*)alloc(N * CDIM * 4);
    float*          ctxB   = (float*)alloc(N * CDIM * 4);
    float*          ctxall = (float*)alloc(N * CDIM * 4);   // later reused as x1
    float*          gates  = (float*)alloc(N * 4 * 4);
    unsigned short* bf1    = (unsigned short*)alloc(N * CDIM * 2); // y_bf -> ctx_all_bf -> ln2_bf
    unsigned short* bf2    = (unsigned short*)alloc(N * CDIM * 2); // x_mod bf16
    unsigned short* hbuf   = (unsigned short*)alloc(N * 512 * 2);  // fc1 hidden bf16
    float* x1 = ctxall;

    // weight prep (transpose -> bf16, pad N where needed)
    k_w2bf_t<<<(320 * 128 + 255) / 256, 256, 0, stream>>>(f_w,   f_wt,   128, 260, 320);
    k_w2bf_t<<<(128 * 128 + 255) / 256, 256, 0, stream>>>(h_w,   h_wt,   128, 128, 128);
    k_w2bf_t<<<(128 * 128 + 255) / 256, 256, 0, stream>>>(proj_w, p_wt,  128, 128, 128);
    k_w2bf_t<<<(512 * 128 + 255) / 256, 256, 0, stream>>>(fc1_w, fc1_wt, 128, 512, 512);
    k_w2bf_t<<<(128 * 512 + 255) / 256, 256, 0, stream>>>(fc2_w, fc2_wt, 512, 128, 128);

    const int MG = (int)(N / 32);            // 6272 m-groups of 32 rows
    const size_t ELEMS = N * CDIM;           // 25,690,112
    const int eBlocks = (int)((ELEMS + 255) / 256);
    const int lnBlocks = (int)((N * 32 + 255) / 256); // one wave per row

    // 1) LN1 -> bf16
    k_ln_bf16<<<lnBlocks, 256, 0, stream>>>(x, ln1_g, ln1_b, bf1, (long long)N);

    // 2) GEMM1: [N,128]x[128,320pad] -> q / ctx / gates
    {
        long long groups = (long long)MG * (20 / 4);
        k_gemm<128, 0><<<(unsigned)((groups + 7) / 8), 256, 0, stream>>>(
            bf1, f_wt, f_b, 20, groups, nullptr, q, ctxA, gates, nullptr);
    }

    // 3) focal levels: dwconv + gelu + gated accumulation
    k_dwconv<<<eBlocks, 256, 0, stream>>>(ctxA, k0w, gates, ctxB, ctxall, 3, 0, 1);
    k_dwconv<<<eBlocks, 256, 0, stream>>>(ctxB, k1w, gates, ctxA, ctxall, 5, 1, 0);
    k_dwconv<<<eBlocks, 256, 0, stream>>>(ctxA, k2w, gates, ctxB, ctxall, 7, 2, 0);

    // 4) global context: mean over (H,W) -> gelu ; then gate3 accumulate + bf16
    k_pool<<<BB * CDIM, 256, 0, stream>>>(ctxB, gbuf);
    k_finalize<<<eBlocks, 256, 0, stream>>>(ctxall, gbuf, gates, bf1);

    // 5) GEMM2: modulator = ctx_all @ h_w + h_b ; x_mod = q * modulator (bf16)
    {
        long long groups = (long long)MG * (8 / 4);
        k_gemm<128, 1><<<(unsigned)((groups + 7) / 8), 256, 0, stream>>>(
            bf1, h_wt, h_b, 8, groups, q, nullptr, nullptr, nullptr, bf2);
    }

    // 6) GEMM3: x1 = shortcut + x_mod @ proj_w + proj_b
    {
        long long groups = (long long)MG * (8 / 4);
        k_gemm<128, 2><<<(unsigned)((groups + 7) / 8), 256, 0, stream>>>(
            bf2, p_wt, proj_b, 8, groups, x, x1, nullptr, nullptr, nullptr);
    }

    // 7) LN2 -> bf16
    k_ln_bf16<<<lnBlocks, 256, 0, stream>>>(x1, ln2_g, ln2_b, bf1, (long long)N);

    // 8) GEMM4: hidden = gelu(ln2 @ fc1_w + fc1_b) (bf16, ld 512)
    {
        long long groups = (long long)MG * (32 / 4);
        k_gemm<128, 3><<<(unsigned)((groups + 7) / 8), 256, 0, stream>>>(
            bf1, fc1_wt, fc1_b, 32, groups, nullptr, nullptr, nullptr, nullptr, hbuf);
    }

    // 9) GEMM5: out = x1 + hidden @ fc2_w + fc2_b   (K = 512)
    {
        long long groups = (long long)MG * (8 / 4);
        k_gemm<512, 4><<<(unsigned)((groups + 7) / 8), 256, 0, stream>>>(
            hbuf, fc2_wt, fc2_b, 8, groups, x1, out, nullptr, nullptr, nullptr);
    }
}
